// BertEncoder_41592463294989
// MI455X (gfx1250) — matrix operand order repaired and checked by reading, not compiled, not run on hardware
//
#include <hip/hip_runtime.h>
#include <hip/hip_bf16.h>

// ---------------------------------------------------------------------------
// BERT encoder (B=8,S=512,HID=768,NH=12,DH=64,INTER=3072,L=4) on gfx1250.
// All GEMM-shaped math via v_wmma_f32_16x16x32_bf16 (fp32 accum).
// ---------------------------------------------------------------------------

typedef unsigned short u16;
typedef __bf16 bf16_t;
typedef bf16_t v16bf __attribute__((ext_vector_type(16)));
typedef float  v8f   __attribute__((ext_vector_type(8)));

#define BB   8
#define SS   512
#define HID  768
#define NHD  12
#define DH   64
#define INTER 3072
#define NLAYERS 4
#define MROWS (BB * SS)          // 4096
#define QKVW (3 * HID)           // 2304

// ---- fp32 -> bf16 via hardware convert (backend picks v_cvt_*_bf16) -------
__device__ __forceinline__ u16 f2bf16u(float f) {
    union { bf16_t h; u16 u; } x;
    x.h = (bf16_t)f;
    return x.u;
}
__device__ __forceinline__ unsigned pack2(float a, float b) {
    return (unsigned)f2bf16u(a) | ((unsigned)f2bf16u(b) << 16);
}
__device__ __forceinline__ uint4 pack8(float4 f0, float4 f1) {
    uint4 p;
    p.x = pack2(f0.x, f0.y); p.y = pack2(f0.z, f0.w);
    p.z = pack2(f1.x, f1.y); p.w = pack2(f1.z, f1.w);
    return p;
}

// ---- WMMA fragment builders (per CDNA5 ISA 16-bit A/B VGPR layouts) --------
// A (16x32, M x K), LDS row-major [M][K], stride in u16 units (mult of 8).
// lane 0..15 : M=lane,    K = {0..7, 16..23}
// lane 16..31: M=lane-16, K = {8..15, 24..31}
__device__ __forceinline__ v16bf load_a_frag(const u16* base, int stride, int lane) {
    int hf = lane >> 4;
    int m  = lane & 15;
    const u16* p = base + m * stride + hf * 8;
    union { uint4 q[2]; v16bf v; } u;
    u.q[0] = *(const uint4*)(p);
    u.q[1] = *(const uint4*)(p + 16);
    return u.v;
}
// B (32x16, K x N) from LDS holding rows of N with K contiguous: [N][K].
// lane 0..15 : N=lane,    K = {0..7, 16..23}
// lane 16..31: N=lane-16, K = {8..15, 24..31}
__device__ __forceinline__ v16bf load_b_frag(const u16* base, int stride, int lane) {
    int hf = lane >> 4;
    int n  = lane & 15;
    const u16* p = base + n * stride + hf * 8;
    union { uint4 q[2]; v16bf v; } u;
    u.q[0] = *(const uint4*)(p);
    u.q[1] = *(const uint4*)(p + 16);
    return u.v;
}
__device__ __forceinline__ v8f wmma_bf16(v16bf a, v16bf b, v8f c) {
    return __builtin_amdgcn_wmma_f32_16x16x32_bf16(
        /*neg_a=*/false, a, /*neg_b=*/false, b,
        /*c_mod=*/(short)0, c, /*reuse_a=*/false, /*reuse_b=*/false);
}

// ---------------------------------------------------------------------------
// GEMM: C[M,N] = A[M,K] @ W[N,K]^T + bias, optional per-row mask multiply.
// 64x64 tile per 256-thread block (8 waves); wave w: rows 16*(w&3),
// cols 32*(w>>2)+{0,16}; K staged 64 at a time in LDS as bf16
// -> 4 WMMAs per wave between barriers.
// ---------------------------------------------------------------------------
__global__ __launch_bounds__(256)
void gemm_wmma_kernel(const float* __restrict__ A, const float* __restrict__ W,
                      const float* __restrict__ bias, const float* __restrict__ rowmask,
                      float* __restrict__ C, int M, int N, int K) {
    __shared__ __align__(16) u16 As[64][72];
    __shared__ __align__(16) u16 Bs[64][72];
    const int tid  = threadIdx.x;
    const int wave = tid >> 5;
    const int lane = tid & 31;
    const int m0 = blockIdx.y * 64;
    const int n0 = blockIdx.x * 64;
    const int rowblk  = wave & 3;   // 16-row sub-block
    const int colpair = wave >> 2;  // 32-col sub-block pair

    const int lr = tid >> 2;        // 0..63 : tile row for cooperative loads
    const int lc = (tid & 3) * 16;  // 0,16,32,48 : K offset

    v8f acc0 = {}; v8f acc1 = {};

    for (int k0 = 0; k0 < K; k0 += 64) {
        { // stage A tile (64x64) as bf16
            const float* src = A + (size_t)(m0 + lr) * K + k0 + lc;
            float4 f0 = *(const float4*)(src);
            float4 f1 = *(const float4*)(src + 4);
            float4 f2 = *(const float4*)(src + 8);
            float4 f3 = *(const float4*)(src + 12);
            *(uint4*)&As[lr][lc]     = pack8(f0, f1);
            *(uint4*)&As[lr][lc + 8] = pack8(f2, f3);
            if (k0 + 64 < K) __builtin_prefetch(src + 64, 0, 1);
        }
        { // stage W tile (64x64) as bf16
            const float* src = W + (size_t)(n0 + lr) * K + k0 + lc;
            float4 f0 = *(const float4*)(src);
            float4 f1 = *(const float4*)(src + 4);
            float4 f2 = *(const float4*)(src + 8);
            float4 f3 = *(const float4*)(src + 12);
            *(uint4*)&Bs[lr][lc]     = pack8(f0, f1);
            *(uint4*)&Bs[lr][lc + 8] = pack8(f2, f3);
            if (k0 + 64 < K) __builtin_prefetch(src + 64, 0, 1);
        }
        __syncthreads();

#pragma unroll
        for (int s = 0; s < 2; ++s) {   // two K=32 chunks
            v16bf afrag = load_a_frag(&As[16 * rowblk][s * 32], 72, lane);
            v16bf b0 = load_b_frag(&Bs[32 * colpair][s * 32], 72, lane);
            v16bf b1 = load_b_frag(&Bs[32 * colpair + 16][s * 32], 72, lane);
            acc0 = wmma_bf16(afrag, b0, acc0);
            acc1 = wmma_bf16(afrag, b1, acc1);
        }
        __syncthreads();
    }

    // Epilogue: C layout — VGPR g: lanes 0-15 M=g, lanes 16-31 M=8+g; N=lane&15
    const int hf = lane >> 4;
    const int nl = lane & 15;
    const int rbase = m0 + 16 * rowblk + hf * 8;
    for (int t = 0; t < 2; ++t) {
        v8f acc = t ? acc1 : acc0;
        const int col = n0 + 32 * colpair + 16 * t + nl;
        const float bv = bias ? bias[col] : 0.0f;
#pragma unroll
        for (int g = 0; g < 8; ++g) {
            const int row = rbase + g;
            float v = acc[g] + bv;
            if (rowmask) v *= rowmask[row];
            C[(size_t)row * N + col] = v;
        }
    }
}

// ---------------------------------------------------------------------------
// Fused attention per (q-block of 32 rows, head, batch).
// Full 32x512 fp32 score block resident in LDS (320 KB WGP LDS makes this
// cheap), KERPLE + mask bias fused into score epilogue, softmax in LDS,
// P@V on WMMA with bf16 probabilities.
// 8 waves tile the 32x64 WMMA output: rowblk = w&1, colblk = w>>1.
// ---------------------------------------------------------------------------
__global__ __launch_bounds__(256)
void attn_wmma_kernel(const float* __restrict__ qkv,   // [B*S][2304]
                      const float* __restrict__ mask,  // [B][S]
                      const float* __restrict__ r1,    // [NH] (layer slice)
                      const float* __restrict__ r2,
                      const float* __restrict__ r3,
                      float* __restrict__ out) {       // [B*S][768]
    __shared__ __align__(16) u16  Qt[32][72];
    __shared__ __align__(16) u16  KVt[64][72];
    __shared__ __align__(16) float Sb[32][528];
    __shared__ __align__(16) u16  Pb[32][520];
    __shared__ float red[32][8];
    __shared__ float rowmax[32];
    __shared__ float rowsum[32];

    const int qb = blockIdx.x;   // 0..15
    const int h  = blockIdx.y;   // 0..11
    const int b  = blockIdx.z;   // 0..7
    const int tid  = threadIdx.x;
    const int wave = tid >> 5;
    const int lane = tid & 31;
    const int rowblk = wave & 1;   // 16-row block of 32
    const int colblk = wave >> 1;  // 16-col block of 64
    const int hf = lane >> 4;
    const int nl = lane & 15;

    const float c1 = fmaxf(r1[h], 1e-7f);
    const float c2 = fmaxf(r2[h], 1e-7f);
    const float c3 = fmaxf(r3[h], 1e-7f);

    { // stage Q block (32x64) as bf16
        const int r = tid >> 3;          // 0..31
        const int c = (tid & 7) * 8;     // 0..56
        const float* src = qkv + ((size_t)(b * SS + qb * 32 + r)) * QKVW + h * DH + c;
        float4 f0 = *(const float4*)(src);
        float4 f1 = *(const float4*)(src + 4);
        *(uint4*)&Qt[r][c] = pack8(f0, f1);
    }

    // ---------------- scores = Q.K^T / 8 + kerple + maskbias ---------------
    for (int kb = 0; kb < 8; ++kb) {
        { // stage K block (64 keys x 64 dims) as bf16, [kpos][d]
            const int r = tid >> 2;         // 0..63
            const int c = (tid & 3) * 16;   // 0,16,32,48
            const float* src = qkv + ((size_t)(b * SS + kb * 64 + r)) * QKVW + HID + h * DH + c;
            float4 f0 = *(const float4*)(src);
            float4 f1 = *(const float4*)(src + 4);
            float4 f2 = *(const float4*)(src + 8);
            float4 f3 = *(const float4*)(src + 12);
            *(uint4*)&KVt[r][c]     = pack8(f0, f1);
            *(uint4*)&KVt[r][c + 8] = pack8(f2, f3);
        }
        __syncthreads();

        v8f acc = {};
#pragma unroll
        for (int s = 0; s < 2; ++s) {  // DH=64 in two K=32 chunks
            v16bf a  = load_a_frag(&Qt[16 * rowblk][s * 32], 72, lane);
            v16bf bf = load_b_frag(&KVt[16 * colblk][s * 32], 72, lane);
            acc = wmma_bf16(a, bf, acc);
        }

        // epilogue -> Sb with fused bias
        const int q0   = 16 * rowblk + hf * 8;
        const int kcol = 16 * colblk + nl;
        const int kpos = kb * 64 + kcol;
        const float mbias = (1.0f - mask[b * SS + kpos]) * -10000.0f;
#pragma unroll
        for (int g = 0; g < 8; ++g) {
            const int qrow = q0 + g;
            const int qpos = qb * 32 + qrow;
            const float rel = fabsf((float)(qpos - kpos));
            const float relp = (rel > 0.0f) ? powf(rel, c3) : 0.0f;
            const float kbias = -c1 * log1pf(c2 * relp);
            Sb[qrow][kpos] = acc[g] * 0.125f + mbias + kbias;
        }
        __syncthreads();
    }

    // ---------------- softmax over rows (fp32 in LDS) ----------------------
    {
        const int r  = tid >> 3;    // 0..31
        const int p  = tid & 7;     // 8 partials per row
        const int c0 = p * 64;
        float mx = -3.0e38f;
        for (int c = 0; c < 64; ++c) mx = fmaxf(mx, Sb[r][c0 + c]);
        red[r][p] = mx;
        __syncthreads();
        if (p == 0) {
            float m2 = red[r][0];
            for (int i = 1; i < 8; ++i) m2 = fmaxf(m2, red[r][i]);
            rowmax[r] = m2;
        }
        __syncthreads();
        const float rm = rowmax[r];
        float sum = 0.0f;
        for (int c = 0; c < 64; ++c) {
            float e = __expf(Sb[r][c0 + c] - rm);
            Sb[r][c0 + c] = e;
            sum += e;
        }
        red[r][p] = sum;
        __syncthreads();
        if (p == 0) {
            float s2 = 0.0f;
            for (int i = 0; i < 8; ++i) s2 += red[r][i];
            rowsum[r] = s2;
        }
        __syncthreads();
        const float inv = 1.0f / rowsum[r];
        for (int c = 0; c < 64; c += 2) {
            float e0 = Sb[r][c0 + c] * inv;
            float e1 = Sb[r][c0 + c + 1] * inv;
            *(unsigned*)&Pb[r][c0 + c] = pack2(e0, e1);
        }
    }
    __syncthreads();

    // ---------------- attn = P @ V (K=512 on WMMA) -------------------------
    v8f accv = {};
    for (int kb = 0; kb < 8; ++kb) {
        { // stage V block transposed: KVt[d][kloc] (bf16) for contiguous B frags
            const int r = tid >> 2;        // kloc 0..63
            const int c = (tid & 3) * 16;  // d base
            const float* src = qkv + ((size_t)(b * SS + kb * 64 + r)) * QKVW + 2 * HID + h * DH + c;
#pragma unroll
            for (int j = 0; j < 16; ++j)
                KVt[c + j][r] = f2bf16u(src[j]);
        }
        __syncthreads();
#pragma unroll
        for (int s = 0; s < 2; ++s) {
            v16bf a  = load_a_frag(&Pb[16 * rowblk][kb * 64 + s * 32], 520, lane);
            v16bf bf = load_b_frag(&KVt[16 * colblk][s * 32], 72, lane);
            accv = wmma_bf16(a, bf, accv);
        }
        __syncthreads();
    }

    { // write attention output in [B,S,NH*DH] layout directly
        const int q0 = 16 * rowblk + hf * 8;
        const int d  = 16 * colblk + nl;
#pragma unroll
        for (int g = 0; g < 8; ++g) {
            const int qpos = qb * 32 + q0 + g;
            out[((size_t)(b * SS + qpos)) * HID + h * DH + d] = accv[g];
        }
    }
}

// ---------------------------------------------------------------------------
// Residual add + LayerNorm (+ optional row-mask). One row per block, 768 cols.
// ---------------------------------------------------------------------------
__global__ __launch_bounds__(256)
void add_ln_kernel(const float* __restrict__ x, const float* __restrict__ res,
                   const float* __restrict__ g, const float* __restrict__ bparm,
                   const float* __restrict__ rowmask, float* __restrict__ out) {
    const int row = blockIdx.x;
    const int tid = threadIdx.x;
    __shared__ float red[256];
    const float* xr = x   + (size_t)row * HID;
    const float* rr = res + (size_t)row * HID;
    float v[3];
    float s = 0.0f;
#pragma unroll
    for (int i = 0; i < 3; ++i) {
        const int c = tid + i * 256;
        v[i] = xr[c] + rr[c];
        s += v[i];
    }
    red[tid] = s; __syncthreads();
    for (int o = 128; o > 0; o >>= 1) { if (tid < o) red[tid] += red[tid + o]; __syncthreads(); }
    const float mu = red[0] * (1.0f / HID);
    __syncthreads();
    float q = 0.0f;
#pragma unroll
    for (int i = 0; i < 3; ++i) { const float d = v[i] - mu; q += d * d; }
    red[tid] = q; __syncthreads();
    for (int o = 128; o > 0; o >>= 1) { if (tid < o) red[tid] += red[tid + o]; __syncthreads(); }
    const float var = red[0] * (1.0f / HID);
    const float inv = rsqrtf(var + 1e-12f);
    const float m = rowmask ? rowmask[row] : 1.0f;
#pragma unroll
    for (int i = 0; i < 3; ++i) {
        const int c = tid + i * 256;
        out[(size_t)row * HID + c] = ((v[i] - mu) * inv * g[c] + bparm[c]) * m;
    }
}

// ---------------------------------------------------------------------------
// GEGLU: x = gelu_exact(g[:, :INTER]) * g[:, INTER:]
// ---------------------------------------------------------------------------
__global__ __launch_bounds__(256)
void geglu_kernel(const float* __restrict__ gl, float* __restrict__ x) {
    const size_t idx = (size_t)blockIdx.x * 256 + threadIdx.x;
    const size_t row = idx / INTER;
    const size_t col = idx - row * INTER;
    const float a = gl[row * (2 * INTER) + col];
    const float b2 = gl[row * (2 * INTER) + INTER + col];
    const float ge = 0.5f * a * (1.0f + erff(a * 0.70710678118654752f));
    x[idx] = ge * b2;
}

// ---------------------------------------------------------------------------
// h = hidden * mask[row]
// ---------------------------------------------------------------------------
__global__ __launch_bounds__(256)
void mask_copy_kernel(const float* __restrict__ in, const float* __restrict__ mask,
                      float* __restrict__ out) {
    const size_t idx = (size_t)blockIdx.x * 256 + threadIdx.x;
    const size_t row = idx / HID;
    out[idx] = in[idx] * mask[row];
}

// ---------------------------------------------------------------------------
extern "C" void kernel_launch(void* const* d_in, const int* in_sizes, int n_in,
                              void* d_out, int out_size, void* d_ws, size_t ws_size,
                              hipStream_t stream) {
    const float* hidden     = (const float*)d_in[0];
    const float* mask       = (const float*)d_in[1];   // [B,S] -> flat [4096]
    const float* Wqkv_w     = (const float*)d_in[2];   // [L,2304,768]
    const float* Wqkv_b     = (const float*)d_in[3];   // [L,2304]
    const float* attn_out_w = (const float*)d_in[4];   // [L,768,768]
    const float* attn_out_b = (const float*)d_in[5];   // [L,768]
    const float* ln1_g      = (const float*)d_in[6];
    const float* ln1_b      = (const float*)d_in[7];
    const float* glu_w      = (const float*)d_in[8];   // [L,6144,768]
    const float* wo_w       = (const float*)d_in[9];   // [L,768,3072]
    const float* wo_b       = (const float*)d_in[10];
    const float* ln2_g      = (const float*)d_in[11];
    const float* ln2_b      = (const float*)d_in[12];
    const float* r1         = (const float*)d_in[13];  // [L,NH]
    const float* r2         = (const float*)d_in[14];
    const float* r3         = (const float*)d_in[15];

    char* ws = (char*)d_ws;
    const size_t TOK = (size_t)MROWS * HID * sizeof(float);       // 12.6 MB
    float* h    = (float*)(ws);                                   // hidden state
    float* ao   = (float*)(ws + TOK);                             // post-LN1
    float* bigA = (float*)(ws + 2 * TOK);                         // qkv / glu   (<=100.7MB)
    float* bigB = (float*)(ws + 2 * TOK + (size_t)MROWS * 2 * INTER * sizeof(float)); // attn / x
    float* go   = (float*)(ws + 2 * TOK + (size_t)MROWS * 2 * INTER * sizeof(float)
                              + (size_t)MROWS * INTER * sizeof(float));               // gemm out

    mask_copy_kernel<<<(MROWS * HID) / 256, 256, 0, stream>>>(hidden, mask, h);

    for (int l = 0; l < NLAYERS; ++l) {
        // QKV projection (masked rows)
        gemm_wmma_kernel<<<dim3(QKVW / 64, MROWS / 64), 256, 0, stream>>>(
            h, Wqkv_w + (size_t)l * QKVW * HID, Wqkv_b + (size_t)l * QKVW,
            mask, bigA, MROWS, QKVW, HID);

        // Fused attention (scores + kerple/mask bias + softmax + P@V)
        attn_wmma_kernel<<<dim3(SS / 32, NHD, BB), 256, 0, stream>>>(
            bigA, mask, r1 + l * NHD, r2 + l * NHD, r3 + l * NHD, bigB);

        // Attention output projection
        gemm_wmma_kernel<<<dim3(HID / 64, MROWS / 64), 256, 0, stream>>>(
            bigB, attn_out_w + (size_t)l * HID * HID, attn_out_b + (size_t)l * HID,
            nullptr, go, MROWS, HID, HID);

        // LN1 with residual h, masked
        add_ln_kernel<<<MROWS, 256, 0, stream>>>(
            go, h, ln1_g + (size_t)l * HID, ln1_b + (size_t)l * HID, mask, ao);

        // GLU projection (no bias)
        gemm_wmma_kernel<<<dim3((2 * INTER) / 64, MROWS / 64), 256, 0, stream>>>(
            ao, glu_w + (size_t)l * 2 * INTER * HID, nullptr, nullptr,
            bigA, MROWS, 2 * INTER, HID);

        // GEGLU activation
        geglu_kernel<<<((size_t)MROWS * INTER) / 256, 256, 0, stream>>>(bigA, bigB);

        // Down projection
        gemm_wmma_kernel<<<dim3(HID / 64, MROWS / 64), 256, 0, stream>>>(
            bigB, wo_w + (size_t)l * HID * INTER, wo_b + (size_t)l * HID,
            nullptr, go, MROWS, HID, INTER);

        // LN2 with residual ao, masked; last layer writes d_out directly
        float* dst = (l == NLAYERS - 1) ? (float*)d_out : h;
        add_ln_kernel<<<MROWS, 256, 0, stream>>>(
            go, ao, ln2_g + (size_t)l * HID, ln2_b + (size_t)l * HID, mask, dst);
    }
}
